// WindowAttention_86260123172918
// MI455X (gfx1250) — matrix operand order repaired and checked by reading, not compiled, not run on hardware
//
#include <hip/hip_runtime.h>
#include <hip/hip_bf16.h>

typedef __attribute__((ext_vector_type(16))) _Float16 v16h;
typedef __attribute__((ext_vector_type(8)))  _Float16 v8h;
typedef __attribute__((ext_vector_type(8)))  float    v8f;

#define NTOK 49
#define DIM 256
#define HEADS 8
#define HD 32
#define NWIN 64
#define NPAD 64

// LDS strides in halfs
#define XS_STRIDE 264   // 256 + 8 pad
#define QK_STRIDE 40    // 32 + 8 pad
#define VT_STRIDE 72    // 64 + 8 pad
#define PS_STRIDE 72

// LDS offsets in halfs
#define XS_OFF 0
#define QS_OFF 16896            // 64*264
#define KS_OFF 37376            // + 8*64*40
#define VT_OFF 57856            // + 8*64*40
#define PS_OFF 76288            // + 8*32*72
#define LDS_HALFS 85504         // + 8*16*72   -> 171008 bytes

static __device__ __forceinline__ v8f vzero() {
    v8f z = {0.f,0.f,0.f,0.f,0.f,0.f,0.f,0.f};
    return z;
}

// Build a 16-half WMMA operand from two 16-byte chunks.
static __device__ __forceinline__ v16h mk16(const _Float16* p0, const _Float16* p1) {
    v8h a = *(const v8h*)p0;
    v8h b = *(const v8h*)p1;
    return __builtin_shufflevector(a, b, 0,1,2,3,4,5,6,7,8,9,10,11,12,13,14,15);
}

static __device__ __forceinline__ v8f wmma16(v16h a, v16h b, v8f c) {
    return __builtin_amdgcn_wmma_f32_16x16x32_f16(false, a, false, b, (short)0, c, false, false);
}

// Repack a row-major f32 weight [K x ncols] into per-WMMA-tile B-operand order:
// block = ntile*8 + ks ; within block: lane*16 + j
// element = W[ks*32 + (lane>>4)*8 + (j&7) + (j>>3)*16][ntile*16 + (lane&15)]  as f16
__global__ void repack_w_kernel(const float* __restrict__ W, _Float16* __restrict__ dst,
                                int ncols, int nblk) {
    int t = blockIdx.x * 256 + threadIdx.x;
    if (t >= nblk * 512) return;
    int blk    = t >> 9;
    int within = t & 511;
    int lane   = within >> 4;
    int j      = within & 15;
    int ntile  = blk >> 3;
    int ks     = blk & 7;
    int row = ks * 32 + ((lane >> 4) << 3) + (j & 7) + ((j >> 3) << 4);
    int col = ntile * 16 + (lane & 15);
    dst[t] = (_Float16)W[row * ncols + col];
}

// bm[w][h][m][n] = bias_table[rel_index[m*49+n]*8 + h] + mask[w][m][n]
__global__ void make_bm_kernel(const float* __restrict__ table, const int* __restrict__ relidx,
                               const float* __restrict__ mask, float* __restrict__ bm) {
    const int per = NTOK * NTOK;
    int t = blockIdx.x * 256 + threadIdx.x;
    if (t >= NWIN * HEADS * per) return;
    int nn = t % per;
    int hh = (t / per) & (HEADS - 1);
    int w  = t / (per * HEADS);
    bm[t] = table[relidx[nn] * HEADS + hh] + mask[w * per + nn];
}

__global__ void __launch_bounds__(256)
win_attn_kernel(const float* __restrict__ x, const _Float16* __restrict__ wq,
                const float* __restrict__ qkvb, const _Float16* __restrict__ wp,
                const float* __restrict__ projb, const float* __restrict__ bm,
                float* __restrict__ out) {
    extern __shared__ _Float16 sh[];
    const int tid  = threadIdx.x;
    const int lane = tid & 31;
    const int h    = tid >> 5;            // wave id == head id
    const int b    = blockIdx.x;          // window instance
    const int w    = b & (NWIN - 1);      // mask window index
    const int lhi  = lane >> 4;
    const int llo  = lane & 15;

    // ---- load x tile: 49x256 f32 -> f16 in LDS, zero-pad rows 49..63 ----
    {
        const float* xb = x + (size_t)b * NTOK * DIM;
        for (int i = tid; i < NPAD * DIM; i += 256) {
            int m = i >> 8, c = i & 255;
            float v = (m < NTOK) ? xb[m * DIM + c] : 0.f;
            sh[XS_OFF + m * XS_STRIDE + c] = (_Float16)v;
        }
    }
    __syncthreads();

    // ---- stage 1: qkv GEMM, wave h computes Q_h, K_h, V_h columns ----
    for (int r = 0; r < 3; ++r) {           // 0=Q 1=K 2=V
        for (int sub = 0; sub < 2; ++sub) {
            const int ntile = r * 16 + 2 * h + sub;      // of 48 column tiles over 768
            v8f acc[4] = { vzero(), vzero(), vzero(), vzero() };
            for (int ks = 0; ks < 8; ++ks) {
                const _Float16* bp = wq + ((size_t)(ntile * 8 + ks) * 512) + lane * 16;
                v16h bmat = mk16(bp, bp + 8);
#pragma unroll
                for (int mt = 0; mt < 4; ++mt) {
                    const _Float16* ar = sh + XS_OFF + (mt * 16 + llo) * XS_STRIDE
                                         + ks * 32 + lhi * 8;
                    acc[mt] = wmma16(mk16(ar, ar + 16), bmat, acc[mt]);
                }
            }
            const float bv = qkvb[ntile * 16 + llo];
            const int dl = sub * 16 + llo;               // head-local column 0..31
#pragma unroll
            for (int mt = 0; mt < 4; ++mt) {
#pragma unroll
                for (int v = 0; v < 8; ++v) {
                    _Float16 hv = (_Float16)(acc[mt][v] + bv);
                    int m = mt * 16 + v + lhi * 8;
                    if (r == 0)      sh[QS_OFF + (h * 64 + m) * QK_STRIDE + dl] = hv;
                    else if (r == 1) sh[KS_OFF + (h * 64 + m) * QK_STRIDE + dl] = hv;
                    else             sh[VT_OFF + (h * HD + dl) * VT_STRIDE + m] = hv; // V^T
                }
            }
        }
    }
    __syncthreads();   // all waves done reading xs -> reuse as Y

    // ---- stage 2: attention for head h (wave-private) ----
    const float scale = 0.1767766953f;     // 1/sqrt(32)
    const float* bmh = bm + ((size_t)(w * HEADS + h)) * NTOK * NTOK;
    const _Float16* Qh = sh + QS_OFF + h * 64 * QK_STRIDE;
    const _Float16* Kh = sh + KS_OFF + h * 64 * QK_STRIDE;
    const _Float16* Vh = sh + VT_OFF + h * HD * VT_STRIDE;
    _Float16* Ph = sh + PS_OFF + h * 16 * PS_STRIDE;
    _Float16* Ys = sh + XS_OFF;

    for (int mt = 0; mt < 4; ++mt) {
        // S row-block = Q(mt) * K^T  (K=32 -> single WMMA K-step)
        v8f s[4];
        {
            const _Float16* aq = Qh + (mt * 16 + llo) * QK_STRIDE + lhi * 16;
            v16h amat = mk16(aq, aq + 8);
#pragma unroll
            for (int nt = 0; nt < 4; ++nt) {
                const _Float16* bp = Kh + (nt * 16 + llo) * QK_STRIDE + lhi * 16;
                s[nt] = wmma16(amat, mk16(bp, bp + 8), vzero());
            }
        }
        // scale + bias + mask, row softmax, write P (f16) to per-wave staging
#pragma unroll
        for (int v = 0; v < 8; ++v) {
            int m = mt * 16 + v + lhi * 8;
#pragma unroll
            for (int nt = 0; nt < 4; ++nt) {
                int n = nt * 16 + llo;
                float sv;
                if (n < NTOK) {
                    float bias = 0.f;
                    if (m < NTOK) bias = bmh[m * NTOK + n];
                    sv = s[nt][v] * scale + bias;
                } else {
                    sv = -1e30f;     // mask padded key columns
                }
                s[nt][v] = sv;
            }
            float mx = fmaxf(fmaxf(s[0][v], s[1][v]), fmaxf(s[2][v], s[3][v]));
            mx = fmaxf(mx, __shfl_xor(mx, 1, 32));
            mx = fmaxf(mx, __shfl_xor(mx, 2, 32));
            mx = fmaxf(mx, __shfl_xor(mx, 4, 32));
            mx = fmaxf(mx, __shfl_xor(mx, 8, 32));
            float sum = 0.f;
#pragma unroll
            for (int nt = 0; nt < 4; ++nt) {
                float e = __expf(s[nt][v] - mx);
                s[nt][v] = e;
                sum += e;
            }
            sum += __shfl_xor(sum, 1, 32);
            sum += __shfl_xor(sum, 2, 32);
            sum += __shfl_xor(sum, 4, 32);
            sum += __shfl_xor(sum, 8, 32);
            float inv = 1.0f / sum;
            int ml = v + lhi * 8;
#pragma unroll
            for (int nt = 0; nt < 4; ++nt)
                Ph[ml * PS_STRIDE + nt * 16 + llo] = (_Float16)(s[nt][v] * inv);
        }
        // O row-block = P * V   (K=64 keys -> 2 K-steps; N=hd=32 -> 2 tiles)
        v8f o0 = vzero(), o1 = vzero();
#pragma unroll
        for (int kk = 0; kk < 2; ++kk) {
            const _Float16* ap = Ph + llo * PS_STRIDE + kk * 32 + lhi * 16;
            v16h amat = mk16(ap, ap + 8);
            const _Float16* b0 = Vh + llo * VT_STRIDE + kk * 32 + lhi * 16;
            const _Float16* b1 = Vh + (16 + llo) * VT_STRIDE + kk * 32 + lhi * 16;
            o0 = wmma16(amat, mk16(b0, b0 + 8), o0);
            o1 = wmma16(amat, mk16(b1, b1 + 8), o1);
        }
#pragma unroll
        for (int v = 0; v < 8; ++v) {
            int m = mt * 16 + v + lhi * 8;
            float y0 = (m < NTOK) ? o0[v] : 0.f;
            float y1 = (m < NTOK) ? o1[v] : 0.f;
            Ys[m * XS_STRIDE + h * 32 + llo]      = (_Float16)y0;
            Ys[m * XS_STRIDE + h * 32 + 16 + llo] = (_Float16)y1;
        }
    }
    __syncthreads();   // Y complete

    // ---- stage 3: proj GEMM, 64 tile jobs over 8 waves ----
    float* ob = out + (size_t)b * NTOK * DIM;
    for (int j = 0; j < 8; ++j) {
        int job = h * 8 + j;
        int nt = job >> 2;        // 0..15 column tiles of 256
        int mt = job & 3;
        v8f acc = vzero();
        for (int ks = 0; ks < 8; ++ks) {
            const _Float16* ar = Ys + (mt * 16 + llo) * XS_STRIDE + ks * 32 + lhi * 8;
            const _Float16* bp = wp + ((size_t)(nt * 8 + ks) * 512) + lane * 16;
            acc = wmma16(mk16(ar, ar + 16), mk16(bp, bp + 8), acc);
        }
        int col = nt * 16 + llo;
        float bv = projb[col];
#pragma unroll
        for (int v = 0; v < 8; ++v) {
            int m = mt * 16 + v + lhi * 8;
            if (m < NTOK) ob[m * DIM + col] = acc[v] + bv;
        }
    }
}

extern "C" void kernel_launch(void* const* d_in, const int* in_sizes, int n_in,
                              void* d_out, int out_size, void* d_ws, size_t ws_size,
                              hipStream_t stream) {
    const float* x          = (const float*)d_in[0];
    const float* mask       = (const float*)d_in[1];
    const float* qkv_w      = (const float*)d_in[2];
    const float* qkv_b      = (const float*)d_in[3];
    const float* proj_w     = (const float*)d_in[4];
    const float* proj_b     = (const float*)d_in[5];
    const float* bias_table = (const float*)d_in[6];
    const int*   rel_index  = (const int*)d_in[7];
    float* out = (float*)d_out;

    char* ws = (char*)d_ws;
    _Float16* wq = (_Float16*)ws;                 // 48*8*512 halfs = 393216 B
    _Float16* wp = (_Float16*)(ws + 393216);      // 16*8*512 halfs = 131072 B
    float*    bm = (float*)(ws + 524288);         // 64*8*49*49 f32 = 4917248 B

    repack_w_kernel<<<768, 256, 0, stream>>>(qkv_w, wq, 3 * DIM, 48 * 8);
    repack_w_kernel<<<256, 256, 0, stream>>>(proj_w, wp, DIM, 16 * 8);
    {
        int total = NWIN * HEADS * NTOK * NTOK;
        make_bm_kernel<<<(total + 255) / 256, 256, 0, stream>>>(bias_table, rel_index, mask, bm);
    }

    int nblocks = in_sizes[0] / (NTOK * DIM);     // 8192 windows
    size_t lds_bytes = (size_t)LDS_HALFS * sizeof(_Float16);   // 171008 B
    win_attn_kernel<<<nblocks, 256, lds_bytes, stream>>>(x, wq, qkv_b, wp, proj_b, bm, out);
}
